// KGAT3_91336774517082
// MI455X (gfx1250) — matrix operand order repaired and checked by reading, not compiled, not run on hardware
//
#include <hip/hip_runtime.h>
#include <hip/hip_bf16.h>
#include <math.h>

#define N_ENT  100000
#define N_RELT 64
#define DDIM   32
#define HHEAD  4
#define HDIM   128
#define NEDGE  500000
#define BQ     8192

typedef __attribute__((ext_vector_type(16))) __bf16 v16bf;
typedef __attribute__((ext_vector_type(8)))  float  v8f;

union V16 {                       // 16 bf16 = 32 bytes = two uint4 vector loads
    v16bf v;
    uint4 u[2];
};

// order-preserving float <-> uint map for atomicMax-based segment max
__device__ __forceinline__ unsigned mapf(float f) {
    unsigned u = __float_as_uint(f);
    return (u & 0x80000000u) ? ~u : (u | 0x80000000u);
}
__device__ __forceinline__ float unmapf(unsigned m) {
    return (m & 0x80000000u) ? __uint_as_float(m & 0x7FFFFFFFu) : __uint_as_float(~m);
}

// ---------------------------------------------------------------------------
// f32 -> bf16 pre-pass (native cvt; tiny vs. the edge phase)
// ---------------------------------------------------------------------------
__global__ void conv_bf16(const float* __restrict__ in, unsigned short* __restrict__ out, int n) {
    int i = blockIdx.x * blockDim.x + threadIdx.x;
    if (i < n) {
        __bf16 b = (__bf16)in[i];
        out[i] = __builtin_bit_cast(unsigned short, b);
    }
}

// ---------------------------------------------------------------------------
// Fused 4-matrix projection GEMM:  O_i = X @ W_i + b_i   (i = q,k,v,skip)
// Xb: [nrows, KDIM] bf16, Wb: [4, KDIM, 128] bf16 (pre-converted),
// O: [nrows, 128] f32. One wave per 16-row block; bf16 WMMA, f32 accumulate.
// A fragment per documented 16-bit A 16x32 layout (K-runs of 8 contiguous ->
// two b128 loads); B fragment N=lane%16, K=element (16 strided 16-bit loads).
// ---------------------------------------------------------------------------
template<int KDIM>
__global__ __launch_bounds__(256)
void gemm4_wmma(const unsigned short* __restrict__ Xb, int nrows,
                const unsigned short* __restrict__ Wb,
                const float* __restrict__ B0, float* __restrict__ O0,
                const float* __restrict__ B1, float* __restrict__ O1,
                const float* __restrict__ B2, float* __restrict__ O2,
                const float* __restrict__ B3, float* __restrict__ O3)
{
    constexpr int KB = KDIM / 32;
    const int lane = threadIdx.x & 31;
    const int wave = threadIdx.x >> 5;
    const int m0   = (blockIdx.x * (blockDim.x >> 5) + wave) << 4;
    if (m0 >= nrows) return;                 // wave-uniform: EXEC all-ones at WMMA
    const int half = lane >> 4;
    const int mrow = m0 + (lane & 15);
    const int ncol = lane & 15;

    // Preload A fragments (shared by all 4 matrices, all 8 column tiles).
    v16bf a[KB];
    #pragma unroll
    for (int kb = 0; kb < KB; ++kb) {
        const unsigned short* xr = Xb + (size_t)mrow * KDIM + kb * 32 + half * 8;
        V16 av;
        av.u[0] = *(const uint4*)(xr);        // K = k0 + half*8 + 0..7
        av.u[1] = *(const uint4*)(xr + 16);   // K = k0 + 16 + half*8 + 0..7
        a[kb] = av.v;
    }

    const float* Bm[4] = {B0, B1, B2, B3};
    float*       Om[4] = {O0, O1, O2, O3};

    for (int mtx = 0; mtx < 4; ++mtx) {
        const unsigned short* W = Wb + (size_t)mtx * KDIM * HDIM;
        const float* Bv = Bm[mtx];
        float*       O  = Om[mtx];
        #pragma unroll
        for (int t = 0; t < 8; ++t) {                       // 8 col tiles of 16
            const int n0 = t * 16;
            const float bias = Bv[n0 + ncol];
            v8f c;
            #pragma unroll
            for (int r = 0; r < 8; ++r) c[r] = bias;        // bias broadcast over rows
            #pragma unroll
            for (int kb = 0; kb < KB; ++kb) {
                const unsigned short* wr = W + (size_t)(kb * 32 + half * 8) * HDIM + n0 + ncol;
                V16 bv;
                #pragma unroll
                for (int i = 0; i < 8; ++i) {
                    bv.v[i]     = __builtin_bit_cast(__bf16, wr[(size_t)i * HDIM]);         // K = k0 + half*8 + i
                    bv.v[8 + i] = __builtin_bit_cast(__bf16, wr[(size_t)(16 + i) * HDIM]);  // K = k0 + 16 + half*8 + i
                }
                c = __builtin_amdgcn_wmma_f32_16x16x32_bf16(
                        false, a[kb], false, bv.v, (short)0, c, false, false);
            }
            #pragma unroll
            for (int r = 0; r < 8; ++r)
                O[(size_t)(m0 + r + half * 8) * HDIM + n0 + ncol] = c[r];
        }
    }
}

// ---------------------------------------------------------------------------
__global__ void init_ms(unsigned* __restrict__ M, float* __restrict__ S, int n) {
    int i = blockIdx.x * blockDim.x + threadIdx.x;
    if (i < n) { M[i] = 0u; S[i] = 0.f; }   // mapped(0) < mapped(any real): -inf proxy
}

// alpha[e,h] = <q[dst,h], k[src,h]> / sqrt(D);  segment max via mapped atomicMax
__global__ void edge_alpha(const int* __restrict__ src, const int* __restrict__ dst,
                           const float* __restrict__ Q, const float* __restrict__ K,
                           float* __restrict__ ALPHA, unsigned* __restrict__ M)
{
    int idx = blockIdx.x * blockDim.x + threadIdx.x;
    if (idx >= NEDGE * HHEAD) return;
    int e = idx >> 2, h = idx & 3;
    int s = src[e], d = dst[e];
    const float4* qr = (const float4*)(Q + (size_t)d * HDIM + h * DDIM);
    const float4* kr = (const float4*)(K + (size_t)s * HDIM + h * DDIM);
    float acc = 0.f;
    #pragma unroll
    for (int i = 0; i < 8; ++i) {
        float4 qa = qr[i], kb = kr[i];
        acc += qa.x * kb.x + qa.y * kb.y + qa.z * kb.z + qa.w * kb.w;
    }
    float alpha = acc * 0.17677669529663687f;   // 1/sqrt(32)
    ALPHA[idx] = alpha;
    atomicMax(&M[d * HHEAD + h], mapf(alpha));
}

// a = exp(alpha - max); segment sum via atomicAdd
__global__ void edge_exp(const int* __restrict__ dst, float* __restrict__ ALPHA,
                         const unsigned* __restrict__ M, float* __restrict__ S)
{
    int idx = blockIdx.x * blockDim.x + threadIdx.x;
    if (idx >= NEDGE * HHEAD) return;
    int e = idx >> 2, h = idx & 3;
    int d = dst[e];
    float m = unmapf(M[d * HHEAD + h]);
    float a = __expf(ALPHA[idx] - m);
    ALPHA[idx] = a;
    atomicAdd(&S[d * HHEAD + h], a);
}

// OUT[dst] += (v[src] + edge_attr) * a/(s+eps); OUT pre-seeded with x@Ws+bs
__global__ void edge_scatter(const int* __restrict__ src, const int* __restrict__ dst,
                             const int* __restrict__ etype, const float* __restrict__ rel,
                             const float* __restrict__ Vb, const float* __restrict__ ALPHA,
                             const float* __restrict__ S, float* __restrict__ OUT)
{
    long long idx = (long long)blockIdx.x * blockDim.x + threadIdx.x;
    if (idx >= (long long)NEDGE * HDIM) return;
    int e  = (int)(idx >> 7);
    int j  = (int)(idx & 127);
    int h  = j >> 5, dd = j & 31;
    int s  = src[e], d = dst[e];
    float w = ALPHA[e * HHEAD + h] / (S[d * HHEAD + h] + 1e-16f);
    int t  = etype[e];
    int t2 = etype[t];                       // faithful double-index quirk of the reference
    float attr = rel[t2 * DDIM + dd];
    float val  = (Vb[(size_t)s * HDIM + j] + attr) * w;
    atomicAdd(&OUT[(size_t)d * HDIM + j], val);
}

// sigmoid( [x[u] ; x[i]] @ Wp + bp ), one wave per pair, shuffle reduction
__global__ void score_kernel(const float* __restrict__ X, const int* __restrict__ ui,
                             const int* __restrict__ ii, const float* __restrict__ Wp,
                             const float* __restrict__ bp, float* __restrict__ out)
{
    int wave = threadIdx.x >> 5;
    int lane = threadIdx.x & 31;
    int b = blockIdx.x * (blockDim.x >> 5) + wave;
    if (b >= BQ) return;
    const float* xu = X + (size_t)ui[b] * HDIM;
    const float* xi = X + (size_t)ii[b] * HDIM;
    float p = 0.f;
    #pragma unroll
    for (int t = 0; t < 4; ++t) {
        int j = lane * 4 + t;
        p += xu[j] * Wp[j] + xi[j] * Wp[HDIM + j];
    }
    #pragma unroll
    for (int off = 16; off > 0; off >>= 1)
        p += __shfl_down(p, off, 32);
    if (lane == 0) out[b] = 1.f / (1.f + __expf(-(p + bp[0])));
}

// ---------------------------------------------------------------------------
extern "C" void kernel_launch(void* const* d_in, const int* in_sizes, int n_in,
                              void* d_out, int out_size, void* d_ws, size_t ws_size,
                              hipStream_t stream)
{
    const int*   edge_index = (const int*)d_in[0];
    const int*   src   = edge_index;
    const int*   dst   = edge_index + NEDGE;
    const int*   etype = (const int*)d_in[1];
    const int*   ui    = (const int*)d_in[2];
    const int*   ii    = (const int*)d_in[3];
    const float* ent   = (const float*)d_in[4];
    const float* rel   = (const float*)d_in[5];
    const float *Wq1 = (const float*)d_in[6],  *bq1 = (const float*)d_in[7];
    const float *Wk1 = (const float*)d_in[8],  *bk1 = (const float*)d_in[9];
    const float *Wv1 = (const float*)d_in[10], *bv1 = (const float*)d_in[11];
    const float *Ws1 = (const float*)d_in[12], *bs1 = (const float*)d_in[13];
    const float *Wq2 = (const float*)d_in[14], *bq2 = (const float*)d_in[15];
    const float *Wk2 = (const float*)d_in[16], *bk2 = (const float*)d_in[17];
    const float *Wv2 = (const float*)d_in[18], *bv2 = (const float*)d_in[19];
    const float *Ws2 = (const float*)d_in[20], *bs2 = (const float*)d_in[21];
    const float *Wp  = (const float*)d_in[22], *bp  = (const float*)d_in[23];

    // workspace carve-up (~294 MB)
    char*  ws = (char*)d_ws;
    const size_t NB = (size_t)N_ENT * HDIM * sizeof(float);     // 51.2 MB
    float*    Q   = (float*)(ws + 0 * NB);
    float*    Kb  = (float*)(ws + 1 * NB);
    float*    Vb  = (float*)(ws + 2 * NB);
    float*    X1  = (float*)(ws + 3 * NB);
    float*    X2  = (float*)(ws + 4 * NB);
    char*     p   = ws + 5 * NB;
    float*    AL  = (float*)p;                 p += (size_t)NEDGE * HHEAD * 4;   // [E,H]
    unsigned* Mx  = (unsigned*)p;              p += (size_t)N_ENT * HHEAD * 4;   // [N,H]
    float*    Sx  = (float*)p;                 p += (size_t)N_ENT * HHEAD * 4;   // [N,H]
    unsigned short* Xbf = (unsigned short*)p;  p += (size_t)N_ENT * HDIM * 2;    // bf16 X (max 25.6 MB)
    unsigned short* Wbf = (unsigned short*)p;  // bf16 weights, 4 * 128*128 * 2 = 128 KB

    const dim3 blk(256);
    const int gemm_blocks = (N_ENT / 16 + 7) / 8;                 // 8 waves/block
    const int nh_blocks   = (N_ENT * HHEAD + 255) / 256;
    const int eh_blocks   = (NEDGE * HHEAD + 255) / 256;
    const int sc_blocks   = (int)(((long long)NEDGE * HDIM + 255) / 256);

    // ---- layer 1 -----------------------------------------------------------
    {
        const int wn = DDIM * HDIM;                                // 4096 per matrix
        conv_bf16<<<(wn + 255) / 256, blk, 0, stream>>>(Wq1, Wbf + 0 * wn, wn);
        conv_bf16<<<(wn + 255) / 256, blk, 0, stream>>>(Wk1, Wbf + 1 * wn, wn);
        conv_bf16<<<(wn + 255) / 256, blk, 0, stream>>>(Wv1, Wbf + 2 * wn, wn);
        conv_bf16<<<(wn + 255) / 256, blk, 0, stream>>>(Ws1, Wbf + 3 * wn, wn);
        const int xn = N_ENT * DDIM;                               // 3.2M
        conv_bf16<<<(xn + 255) / 256, blk, 0, stream>>>(ent, Xbf, xn);
    }
    gemm4_wmma<DDIM><<<gemm_blocks, blk, 0, stream>>>(Xbf, N_ENT, Wbf,
        bq1, Q,  bk1, Kb,  bv1, Vb,  bs1, X1);
    init_ms<<<nh_blocks, blk, 0, stream>>>(Mx, Sx, N_ENT * HHEAD);
    edge_alpha<<<eh_blocks, blk, 0, stream>>>(src, dst, Q, Kb, AL, Mx);
    edge_exp<<<eh_blocks, blk, 0, stream>>>(dst, AL, Mx, Sx);
    edge_scatter<<<sc_blocks, blk, 0, stream>>>(src, dst, etype, rel, Vb, AL, Sx, X1);

    // ---- layer 2 -----------------------------------------------------------
    {
        const int wn = HDIM * HDIM;                                // 16384 per matrix
        conv_bf16<<<(wn + 255) / 256, blk, 0, stream>>>(Wq2, Wbf + 0 * wn, wn);
        conv_bf16<<<(wn + 255) / 256, blk, 0, stream>>>(Wk2, Wbf + 1 * wn, wn);
        conv_bf16<<<(wn + 255) / 256, blk, 0, stream>>>(Wv2, Wbf + 2 * wn, wn);
        conv_bf16<<<(wn + 255) / 256, blk, 0, stream>>>(Ws2, Wbf + 3 * wn, wn);
        const int xn = N_ENT * HDIM;                               // 12.8M
        conv_bf16<<<(xn + 255) / 256, blk, 0, stream>>>(X1, Xbf, xn);
    }
    gemm4_wmma<HDIM><<<gemm_blocks, blk, 0, stream>>>(Xbf, N_ENT, Wbf,
        bq2, Q,  bk2, Kb,  bv2, Vb,  bs2, X2);
    init_ms<<<nh_blocks, blk, 0, stream>>>(Mx, Sx, N_ENT * HHEAD);
    edge_alpha<<<eh_blocks, blk, 0, stream>>>(src, dst, Q, Kb, AL, Mx);
    edge_exp<<<eh_blocks, blk, 0, stream>>>(dst, AL, Mx, Sx);
    edge_scatter<<<sc_blocks, blk, 0, stream>>>(src, dst, etype, rel, Vb, AL, Sx, X2);

    // ---- scorer ------------------------------------------------------------
    score_kernel<<<BQ / 8, blk, 0, stream>>>(X2, ui, ii, Wp, bp, (float*)d_out);
}
